// HeadEdge_16801912062758
// MI455X (gfx1250) — matrix operand order repaired and checked by reading, not compiled, not run on hardware
//
#include <hip/hip_runtime.h>
#include <math.h>

typedef __attribute__((ext_vector_type(16))) _Float16 v16h;
typedef __attribute__((ext_vector_type(8)))  _Float16 v8h;
typedef __attribute__((ext_vector_type(8)))  float    v8f;
typedef __attribute__((ext_vector_type(4)))  unsigned int v4u;
typedef __attribute__((ext_vector_type(8)))  int          v8i;
typedef __attribute__((ext_vector_type(4)))  int          v4i;

#define SP  512
#define CIN 256
#define D1  128
#define D2  64
#define D3  32
#define CHUNK_ELTS (128 * D1)   // one pass worth of H rows: 128 x 128 f32 = 64 KB

// ---------------------------------------------------------------------------
// Kernel A: H[n*512+j][c] = sum_k points[n][k][j] * W0[k][c]
// (first linear layer factored out of the pairwise sum: (pi+pj)W0 = piW0+pjW0)
// Also zeroes the global loss accumulators.
// ---------------------------------------------------------------------------
__global__ __launch_bounds__(128)
void h_kernel(const float* __restrict__ points, const float* __restrict__ W0,
              float* __restrict__ H, float* __restrict__ accum)
{
    __shared__ float col[CIN];
    const int blk = blockIdx.x;          // n*512 + j
    const int n = blk >> 9, j = blk & 511;
    const float* pp = points + (size_t)n * CIN * SP + j;
    for (int k = threadIdx.x; k < CIN; k += 128) col[k] = pp[(size_t)k * SP];
    if (blk == 0 && threadIdx.x < 2) accum[threadIdx.x] = 0.0f;
    __syncthreads();
    const int c = threadIdx.x;           // 0..127
    float acc = 0.0f;
    for (int k = 0; k < CIN; ++k) acc = fmaf(col[k], W0[k * D1 + c], acc);
    H[(size_t)blk * D1 + c] = acc;
}

// ---------------------------------------------------------------------------
// TDM: async 64 KB 1-D tile load (global -> LDS), tracked on TENSORcnt.
// D# group0: [0]=count=1, [1]=lds_addr, [2]=ga[31:0], [3]=ga[56:32]|type(2)<<30
// D# group1: data_size=4B, tensor_dim0=tile_dim0=16384, stride0=16384,
//            higher dims unused.
// ---------------------------------------------------------------------------
__device__ __forceinline__ void tdm_load_chunk(const float* gsrc, void* ldst)
{
    const unsigned long long ga = (unsigned long long)(uintptr_t)gsrc;
    const unsigned int la = (unsigned int)(uintptr_t)ldst;  // LDS byte offset

    v4u g0;
    g0[0] = 1u;                                              // count=1
    g0[1] = la;                                              // lds_addr
    g0[2] = (unsigned int)ga;                                // global_addr lo
    g0[3] = (unsigned int)((ga >> 32) & 0x01FFFFFFu) | (2u << 30); // hi|type

    v8i g1;
    g1[0] = (int)(2u << 16);              // wg_mask=0, data_size=4B
    g1[1] = (int)(16384u << 16);          // tensor_dim0 lo16 (=16384)
    g1[2] = (int)(1u << 16);              // tensor_dim0 hi16=0 | tensor_dim1=1
    g1[3] = (int)(16384u << 16);          // tensor_dim1 hi=0  | tile_dim0=16384
    g1[4] = 0;                            // tile_dim1/2 unused
    g1[5] = (int)16384u;                  // tensor_dim0_stride lo32
    g1[6] = 0;
    g1[7] = 0;

    v4i z4 = {0, 0, 0, 0};
#if __clang_major__ >= 23
    v8i z8 = {0, 0, 0, 0, 0, 0, 0, 0};
    __builtin_amdgcn_tensor_load_to_lds(g0, g1, z4, z4, z8, 0);
#else
    __builtin_amdgcn_tensor_load_to_lds(g0, g1, z4, z4, 0);
#endif
}

// ---------------------------------------------------------------------------
// Kernel B: per (n,i) block; per wave: 16-row pair tiles through
// relu(H_i+H_j+b0) -> [WMMA f16 128->64] -> relu -> [WMMA f16 64->32] -> relu
// -> 32-MAC dot (->1) -> pred + masked BCE accumulation.
// H pass-chunks (64 KB) are staged into LDS by the Tensor Data Mover,
// double-buffered so the DMA of pass p+1 overlaps the WMMA work of pass p.
// ---------------------------------------------------------------------------
__global__ __launch_bounds__(256)
void mlp_kernel(const float* __restrict__ H,
                const float* __restrict__ b0v,
                const float* __restrict__ W1, const float* __restrict__ b1v,
                const float* __restrict__ W2, const float* __restrict__ b2v,
                const float* __restrict__ W3, const float* __restrict__ b3v,
                const float* __restrict__ heat,
                const float* __restrict__ mask,
                const float* __restrict__ pmask,
                float* __restrict__ pred,
                float* __restrict__ accum)
{
    __shared__ float Hbuf[2][CHUNK_ELTS];  // 2 x 64 KB TDM staging buffers
    __shared__ _Float16 W1t[D2][D1];       // transposed: [col][k], 16 KB
    __shared__ _Float16 W2t[D3][D2];       // transposed: [col][k], 4 KB
    __shared__ float HiB0[D1];             // H[n,i,:] + b0
    __shared__ float B1s[D2], B2s[D3], W3s[D3];
    __shared__ float b3s, redN, redD;
    __shared__ _Float16 x2s[8][16 * D2];   // per-wave layer1 output, 16 KB
    __shared__ float    x3s[8][16 * D3];   // per-wave layer2 output, 16 KB

    const int tid = threadIdx.x;
    const int blk = blockIdx.x;            // n*512 + i
    const int n = blk >> 9, i = blk & 511;
    const float* Hn = H + (size_t)n * SP * D1;

    // wave 0 kicks off the DMA for pass 0 while everyone stages weights
    if (tid < 32) tdm_load_chunk(Hn, &Hbuf[0][0]);

    // ---- prologue: stage weights (f16, transposed), biases, H_i ----
    for (int t = tid; t < D1 * D2; t += 256) {
        int k = t / D2, c = t % D2;
        W1t[c][k] = (_Float16)W1[t];
    }
    for (int t = tid; t < D2 * D3; t += 256) {
        int k = t / D3, c = t % D3;
        W2t[c][k] = (_Float16)W2[t];
    }
    if (tid < D1) HiB0[tid] = H[(size_t)blk * D1 + tid] + b0v[tid];
    if (tid < D2) B1s[tid] = b1v[tid];
    if (tid < D3) { B2s[tid] = b2v[tid]; W3s[tid] = W3[tid]; }
    if (tid == 0) { b3s = b3v[0]; redN = 0.0f; redD = 0.0f; }

    if (tid < 32) __builtin_amdgcn_s_wait_tensorcnt((short)0);
    __syncthreads();   // weights staged AND Hbuf[0] DMA complete

    const int wave = tid >> 5;
    const int lane = tid & 31;
    const int half = lane >> 4;
    const int r    = lane & 15;

    // ---- B fragments, resident in VGPRs for the wave's lifetime ----
    // 16-bit B 32x16 layout: lanes 0-15 hold K=0..7,16..23 of column (lane&15),
    // lanes 16-31 hold K=8..15,24..31.  W1t rows are contiguous in K.
    v16h B1f[4][4];
    #pragma unroll
    for (int nb = 0; nb < 4; ++nb)
        #pragma unroll
        for (int kc = 0; kc < 4; ++kc)
            B1f[nb][kc] = __builtin_shufflevector(*(const v8h*)&W1t[nb * 16 + r][kc * 32 + half * 8], *(const v8h*)&W1t[nb * 16 + r][kc * 32 + half * 8 + 16], 0, 1, 2, 3, 4, 5, 6, 7, 8, 9, 10, 11, 12, 13, 14, 15);
    v16h B2f[2][2];
    #pragma unroll
    for (int nb = 0; nb < 2; ++nb)
        #pragma unroll
        for (int kc = 0; kc < 2; ++kc)
            B2f[nb][kc] = __builtin_shufflevector(*(const v8h*)&W2t[nb * 16 + r][kc * 32 + half * 8], *(const v8h*)&W2t[nb * 16 + r][kc * 32 + half * 8 + 16], 0, 1, 2, 3, 4, 5, 6, 7, 8, 9, 10, 11, 12, 13, 14, 15);

    float pn = 0.0f, pd = 0.0f;

    for (int pass = 0; pass < 4; ++pass) {
        // prefetch next pass's 64 KB H chunk into the other buffer
        // (its last readers synced at the barrier that opened this pass)
        if (tid < 32 && pass < 3)
            tdm_load_chunk(Hn + (size_t)(pass + 1) * CHUNK_ELTS,
                           &Hbuf[(pass + 1) & 1][0]);

        const int j0 = pass * 128 + wave * 16;
        const float4* Hj4 =
            (const float4*)&Hbuf[pass & 1][(wave * 16 + r) * D1];

        // ---- layer-1 A fragments: relu(H_i + H_j + b0) in f16 A layout ----
        // 16-bit A 16x32 layout: lane holds row M=lane&15; lanes 0-15 get
        // K = kb..kb+7 (lo 8 halves) and kb+16..kb+23 (hi 8); lanes 16-31 +8.
        v16h A1[4];
        #pragma unroll
        for (int kc = 0; kc < 4; ++kc) {
            const int kb = kc * 32 + half * 8;   // multiple of 8
            float vals[16];
            *(float4*)&vals[0]  = Hj4[(kb >> 2) + 0];
            *(float4*)&vals[4]  = Hj4[(kb >> 2) + 1];
            *(float4*)&vals[8]  = Hj4[(kb >> 2) + 4];
            *(float4*)&vals[12] = Hj4[(kb >> 2) + 5];
            v16h f;
            #pragma unroll
            for (int e = 0; e < 8; ++e)
                f[e] = (_Float16)fmaxf(vals[e] + HiB0[kb + e], 0.0f);
            #pragma unroll
            for (int e = 8; e < 16; ++e)
                f[e] = (_Float16)fmaxf(vals[e] + HiB0[kb + 8 + e], 0.0f);
            A1[kc] = f;
        }

        // ---- layer 1: (16x128) x (128x64) via 16 WMMAs ----
        #pragma unroll
        for (int nb = 0; nb < 4; ++nb) {
            const float bc = B1s[nb * 16 + r];
            v8f acc = {bc, bc, bc, bc, bc, bc, bc, bc};
            #pragma unroll
            for (int kc = 0; kc < 4; ++kc)
                acc = __builtin_amdgcn_wmma_f32_16x16x32_f16(
                    false, A1[kc], false, B1f[nb][kc], (short)0, acc, false, false);
            // D layout -> row-major LDS scratch (transpose for next A)
            #pragma unroll
            for (int e = 0; e < 8; ++e)
                x2s[wave][(e + half * 8) * D2 + nb * 16 + r] =
                    (_Float16)fmaxf(acc[e], 0.0f);
        }
        asm volatile("s_wait_dscnt 0" ::: "memory");
        __builtin_amdgcn_wave_barrier();

        // ---- layer-2 A fragments from LDS scratch ----
        v16h A2[2];
        #pragma unroll
        for (int kc = 0; kc < 2; ++kc) {
            const int kb = kc * 32 + half * 8;
            v8h lo = *(const v8h*)&x2s[wave][r * D2 + kb];
            v8h hi = *(const v8h*)&x2s[wave][r * D2 + kb + 16];
            v16h f;
            #pragma unroll
            for (int e = 0; e < 8; ++e) { f[e] = lo[e]; f[8 + e] = hi[e]; }
            A2[kc] = f;
        }

        // ---- layer 2: (16x64) x (64x32) via 4 WMMAs ----
        #pragma unroll
        for (int nb = 0; nb < 2; ++nb) {
            const float bc = B2s[nb * 16 + r];
            v8f acc = {bc, bc, bc, bc, bc, bc, bc, bc};
            #pragma unroll
            for (int kc = 0; kc < 2; ++kc)
                acc = __builtin_amdgcn_wmma_f32_16x16x32_f16(
                    false, A2[kc], false, B2f[nb][kc], (short)0, acc, false, false);
            #pragma unroll
            for (int e = 0; e < 8; ++e)
                x3s[wave][(e + half * 8) * D3 + nb * 16 + r] = fmaxf(acc[e], 0.0f);
        }
        asm volatile("s_wait_dscnt 0" ::: "memory");
        __builtin_amdgcn_wave_barrier();

        // ---- layer 3 (32->1) + masked BCE, one row per lane 0..15 ----
        if (lane < 16) {
            float z = b3s;
            #pragma unroll
            for (int c = 0; c < D3; ++c)
                z = fmaf(x3s[wave][r * D3 + c], W3s[c], z);
            const size_t idx = ((size_t)n * SP + i) * SP + (size_t)(j0 + r);
            pred[idx] = z;
            const float t  = heat[idx];
            const float mk = mask[idx];
            const float pm = pmask[idx];
            const float l  = fmaxf(z, 0.0f) - z * t + log1pf(expf(-fabsf(z)));
            pn += l * mk * pm;
            pd += pm;
        }

        // make pass p+1's chunk visible to everyone before it is read
        if (pass < 3) {
            if (tid < 32) __builtin_amdgcn_s_wait_tensorcnt((short)0);
            __syncthreads();
        }
    }

    // ---- block reduction -> one global atomic pair per block ----
    atomicAdd(&redN, pn);
    atomicAdd(&redD, pd);
    __syncthreads();
    if (tid == 0) {
        atomicAdd(&accum[0], redN);
        atomicAdd(&accum[1], redD);
    }
}

// ---------------------------------------------------------------------------
// Kernel C: loss = num / den
// ---------------------------------------------------------------------------
__global__ void finalize_kernel(const float* __restrict__ accum,
                                float* __restrict__ out_loss)
{
    if (threadIdx.x == 0 && blockIdx.x == 0)
        out_loss[0] = accum[0] / accum[1];
}

// ---------------------------------------------------------------------------
extern "C" void kernel_launch(void* const* d_in, const int* in_sizes, int n_in,
                              void* d_out, int out_size, void* d_ws, size_t ws_size,
                              hipStream_t stream)
{
    const float* points = (const float*)d_in[0];
    const float* heat   = (const float*)d_in[1];
    const float* mask   = (const float*)d_in[2];
    const float* pmask  = (const float*)d_in[3];
    const float* W0 = (const float*)d_in[4];
    const float* b0 = (const float*)d_in[5];
    const float* W1 = (const float*)d_in[6];
    const float* b1 = (const float*)d_in[7];
    const float* W2 = (const float*)d_in[8];
    const float* b2 = (const float*)d_in[9];
    const float* W3 = (const float*)d_in[10];
    const float* b3 = (const float*)d_in[11];

    float* accum = (float*)d_ws;                 // [0]=num, [1]=den
    float* Hws   = (float*)d_ws + 64;            // (2*512) x 128 fp32 = 512 KB
    float* pred  = (float*)d_out;                // 2*512*512 floats
    float* lossp = (float*)d_out + 2 * SP * SP;  // final scalar

    h_kernel<<<2 * SP, 128, 0, stream>>>(points, W0, Hws, accum);
    mlp_kernel<<<2 * SP, 256, 0, stream>>>(Hws, b0, W1, b1, W2, b2, W3, b3,
                                           heat, mask, pmask, pred, accum);
    finalize_kernel<<<1, 1, 0, stream>>>(accum, lossp);
}
